// SelectiveAttention_82351702933529
// MI455X (gfx1250) — matrix operand order repaired and checked by reading, not compiled, not run on hardware
//
#include <hip/hip_runtime.h>
#include <hip/hip_bf16.h>

typedef __attribute__((ext_vector_type(16))) __bf16 v16bf;
typedef __attribute__((ext_vector_type(8)))  __bf16 v8bf;
typedef __attribute__((ext_vector_type(8)))  float  v8f;
typedef __attribute__((ext_vector_type(4)))  float  v4f;
typedef __attribute__((ext_vector_type(4)))  unsigned v4u;
typedef __attribute__((ext_vector_type(8)))  int      v8i;
typedef __attribute__((ext_vector_type(4)))  int      v4i;

#define WMMA_BF16(a, b, c) \
  __builtin_amdgcn_wmma_f32_16x16x32_bf16(false, (a), false, (b), (short)0, (c), false, false)

static __device__ __forceinline__ __bf16 f2bf(float x) { return (__bf16)x; }

static __device__ __forceinline__ v16bf cat16(v8bf lo, v8bf hi) {
  return __builtin_shufflevector(lo, hi, 0, 1, 2, 3, 4, 5, 6, 7,
                                         8, 9, 10, 11, 12, 13, 14, 15);
}

static __device__ __forceinline__ v16bf ld_b_op(const __bf16* p) {
  v8bf lo = *(const v8bf*)(p);
  v8bf hi = *(const v8bf*)(p + 16);
  return cat16(lo, hi);
}

constexpr int Bn  = 4;
constexpr int Sn  = 4096;
constexpr int Dn  = 512;
constexpr int Dkn = 64;
constexpr int KWIN = 320;                 // kv band window kept in LDS
constexpr int KPAD = 72;                  // padded LDS row stride (elements)

// ---------------------------------------------------------------------------
// Kernel 0: build transposed bf16 weights.
// ---------------------------------------------------------------------------
__global__ void sa_prep(const float* __restrict__ Wi, const float* __restrict__ Wo,
                        __bf16* __restrict__ WiT, __bf16* __restrict__ WoT) {
  int i = blockIdx.x * 256 + threadIdx.x;
  if (i < Dn * Dkn) {
    const int k1 = i / Dkn, n1 = i % Dkn;        // Wi is [512][64]
    WiT[(size_t)n1 * Dn + k1] = f2bf(Wi[i]);
    const int k2 = i / Dn, n2 = i % Dn;          // Wo is [64][512]
    WoT[(size_t)n2 * Dkn + k2] = f2bf(Wo[i]);
  }
}

// ---------------------------------------------------------------------------
// Kernel 1: P = X @ Wi + bi -> bf16. Two 16x64 tiles per wave (B-op reuse).
// grid.y: 0 = query -> qB; 1 = value -> kvB + kvT.
// ---------------------------------------------------------------------------
__global__ void __launch_bounds__(128)
sa_proj(const float* __restrict__ query, const float* __restrict__ value,
        const __bf16* __restrict__ WiT, const float* __restrict__ bi,
        __bf16* __restrict__ qB, __bf16* __restrict__ kvB,
        __bf16* __restrict__ kvT) {
  const int lane = threadIdx.x & 31;
  const int wave = threadIdx.x >> 5;
  const int tile = blockIdx.x * 8 + wave * 2;    // 2 consecutive tiles per wave
  const int row0 = tile * 16;                    // 32 rows per wave
  const int m = lane & 15;
  const int g = lane >> 4;

  const float* X = blockIdx.y ? value : query;
  __bf16*      P = blockIdx.y ? kvB : qB;

  v8f acc[2][4] = {};
  const float* xr0 = X + (size_t)(row0 + m) * Dn;
  const float* xr1 = X + (size_t)(row0 + 16 + m) * Dn;

  for (int kb = 0; kb < Dn; kb += 32) {
    v16bf a0, a1;
    {
      const float* xp = xr0 + kb + g * 8;
      v4f x0 = *(const v4f*)(xp),      x1 = *(const v4f*)(xp + 4);
      v4f x2 = *(const v4f*)(xp + 16), x3 = *(const v4f*)(xp + 20);
#pragma unroll
      for (int h = 0; h < 4; ++h) {
        a0[h] = f2bf(x0[h]); a0[h + 4] = f2bf(x1[h]);
        a0[h + 8] = f2bf(x2[h]); a0[h + 12] = f2bf(x3[h]);
      }
    }
    {
      const float* xp = xr1 + kb + g * 8;
      v4f x0 = *(const v4f*)(xp),      x1 = *(const v4f*)(xp + 4);
      v4f x2 = *(const v4f*)(xp + 16), x3 = *(const v4f*)(xp + 20);
#pragma unroll
      for (int h = 0; h < 4; ++h) {
        a1[h] = f2bf(x0[h]); a1[h + 4] = f2bf(x1[h]);
        a1[h + 8] = f2bf(x2[h]); a1[h + 12] = f2bf(x3[h]);
      }
    }
#pragma unroll
    for (int nt = 0; nt < 4; ++nt) {
      const v16bf bm = ld_b_op(WiT + (size_t)(nt * 16 + m) * Dn + kb + g * 8);
      acc[0][nt] = WMMA_BF16(a0, bm, acc[0][nt]);   // one B load feeds 2 wmma
      acc[1][nt] = WMMA_BF16(a1, bm, acc[1][nt]);
    }
  }

#pragma unroll
  for (int u = 0; u < 2; ++u) {
    const int r0 = row0 + u * 16;
    const int bt = r0 >> 12;                  // batch (Sn = 4096)
    const int s0 = r0 & (Sn - 1);
#pragma unroll
    for (int nt = 0; nt < 4; ++nt) {
      const float bv = bi[nt * 16 + m];
      v8bf pk;
#pragma unroll
      for (int r = 0; r < 8; ++r) pk[r] = f2bf(acc[u][nt][r] + bv);
#pragma unroll
      for (int r = 0; r < 8; ++r)
        P[(size_t)(r0 + r + 8 * g) * Dkn + nt * 16 + m] = pk[r];
      if (blockIdx.y)   // transposed copy: kvT[b][d][s], 8 contiguous along s
        *(v8bf*)(kvT + ((size_t)bt * Dkn + nt * 16 + m) * Sn + s0 + 8 * g) = pk;
    }
  }
}

// ---------------------------------------------------------------------------
// Kernel 2: fused banded attention + output projection.
// Block = 4 waves = 64 consecutive queries sharing a 320-key kv window that
// the Tensor Data Mover stages into LDS (padded rows, bank-conflict-free).
// ---------------------------------------------------------------------------
__global__ void __launch_bounds__(128)
sa_attn(const __bf16* __restrict__ qB, const __bf16* __restrict__ kvB,
        const __bf16* __restrict__ kvT, const __bf16* __restrict__ WoT,
        const float* __restrict__ bo, const int* __restrict__ band_ptr,
        float* __restrict__ out) {
  __shared__ __bf16 kvLds[KWIN * KPAD];     // 45KB padded kv band
  __shared__ float  scr[4][16][68];         // per-wave transpose scratch

  const int lane  = threadIdx.x & 31;
  const int wave  = threadIdx.x >> 5;
  const int b     = blockIdx.x >> 6;        // 64 blocks per batch
  const int q0blk = (blockIdx.x & 63) * 64; // first query of block (mult of 64)
  const int q0    = q0blk + wave * 16;
  const int band  = band_ptr[0];
  const int m = lane & 15;
  const int g = lane >> 4;
  const float sscale = 0.125f;              // 1/sqrt(64)

  const __bf16* kv_b  = kvB + (size_t)b * Sn * Dkn;
  const __bf16* kvT_b = kvT + (size_t)b * Dkn * Sn;

  // --- stage the shared kv window into LDS ---
  int kstart = q0blk - 128;
  if (kstart < 0) kstart = 0;
  if (kstart > Sn - KWIN) kstart = Sn - KWIN;

#if __has_builtin(__builtin_amdgcn_tensor_load_to_lds)
  if (wave == 0) {
    const unsigned lds_off = (unsigned)(size_t)(void*)kvLds;  // LDS byte offset
    const unsigned long long ga =
        (unsigned long long)(const void*)(kv_b + (size_t)kstart * Dkn);
    v4u g0 = {};
    g0[0] = 1u;                                   // count = 1 valid descriptor
    g0[1] = lds_off;                              // LDS dest
    g0[2] = (unsigned)ga;                         // global addr [31:0]
    g0[3] = (unsigned)((ga >> 32) & 0x01FFFFFFu) | (2u << 30);  // addr hi | type=2
    v8i g1 = {};
    // data_size=2B | pad_enable | pad_interval=32dw (code 4) | pad_amount=4dw (code 3)
    g1[0] = (int)((1u << 16) | (1u << 20) | (4u << 22) | (3u << 25));
    g1[1] = (int)(64u << 16);                     // tensor_dim0 = 64 (lo16)
    g1[2] = (int)(((unsigned)(Sn - kstart) & 0xFFFFu) << 16);   // tensor_dim1 lo16
    g1[3] = (int)(64u << 16);                     // tile_dim0 = 64
    g1[4] = (int)(KWIN);                          // tile_dim1 = 320, tile_dim2 = 0
    g1[5] = 64;                                   // tensor_dim0_stride = 64
    v4i z4 = {};
#if defined(__clang_major__) && (__clang_major__ >= 23)
    v8i z8 = {};
    __builtin_amdgcn_tensor_load_to_lds(g0, g1, z4, z4, z8, 0);
#else
    __builtin_amdgcn_tensor_load_to_lds(g0, g1, z4, z4, 0);
#endif
    __builtin_amdgcn_s_wait_tensorcnt(0);
  }
#else
  // Fallback: cooperative vector copy with the same padded layout.
  for (int idx = threadIdx.x; idx < KWIN * 8; idx += 128) {
    const int row = idx >> 3, c8 = (idx & 7) * 8;
    *(v8bf*)(kvLds + (size_t)row * KPAD + c8) =
        *(const v8bf*)(kv_b + (size_t)(kstart + row) * Dkn + c8);
  }
#endif
  __syncthreads();

  // --- q A-operands for K = 0..31 and 32..63 ---
  v16bf qa[2];
  {
    const __bf16* qr = qB + ((size_t)b * Sn + q0 + m) * Dkn;
#pragma unroll
    for (int p = 0; p < 2; ++p) {
      v8bf lo = *(const v8bf*)(qr + p * 32 + g * 8);
      v8bf hi = *(const v8bf*)(qr + p * 32 + g * 8 + 16);
      qa[p] = cat16(lo, hi);
    }
  }

  v8f at[4] = {};

  const int kbase = (q0 - band) & ~31;      // floor to 32 (ok for negatives)
  const int nblk  = (q0 + 15 + band - kbase) / 32 + 1;

  for (int jb = 0; jb < nblk; ++jb) {
    const int k0 = kbase + jb * 32;
    if (k0 + 32 <= 0 || k0 >= Sn) continue;
    {
      const int kn = k0 + 32;               // prefetch next kvT block
      if (kn >= 0 && kn < Sn) __builtin_prefetch(kvT_b + (size_t)m * Sn + kn, 0, 1);
    }
    const bool inb = (k0 >= 0) && (k0 + 32 <= Sn);

    // ---- scores: two 16x16 tiles, K = 64, kv rows from LDS ----
#pragma unroll
    for (int t = 0; t < 2; ++t) {
      v8f s = {};
      const int key  = k0 + t * 16 + m;
      const int keyc = key < 0 ? 0 : (key >= Sn ? Sn - 1 : key);
      const __bf16* kr = kvLds + (size_t)(keyc - kstart) * KPAD + g * 8;
#pragma unroll
      for (int p = 0; p < 2; ++p)
        s = WMMA_BF16(qa[p], ld_b_op(kr + p * 32), s);
#pragma unroll
      for (int r = 0; r < 8; ++r) {
        const int qrow = q0 + r + 8 * g;
        const int dd   = key - qrow;
        const bool ok  = (key >= 0) && (key < Sn) && (dd <= band) && (dd >= -band);
        scr[wave][r + 8 * g][t * 16 + m] = ok ? s[r] * sscale : 0.0f;
      }
    }

    // ---- scores 16x32 as bf16 A-operand (C->A transpose via LDS) ----
    v16bf sa;
    {
      const float* srow = &scr[wave][m][g * 8];
      v4f s0 = *(const v4f*)(srow),      s1 = *(const v4f*)(srow + 4);
      v4f s2 = *(const v4f*)(srow + 16), s3 = *(const v4f*)(srow + 20);
#pragma unroll
      for (int h = 0; h < 4; ++h) {
        sa[h] = f2bf(s0[h]); sa[h + 4] = f2bf(s1[h]);
        sa[h + 8] = f2bf(s2[h]); sa[h + 12] = f2bf(s3[h]);
      }
    }

    // ---- attended += scores @ kv (K = 32 keys) from transposed copy ----
#pragma unroll
    for (int nt = 0; nt < 4; ++nt) {
      v16bf kvop;
      const __bf16* kd = kvT_b + (size_t)(nt * 16 + m) * Sn;
      if (inb) {
        kvop = ld_b_op(kd + k0 + g * 8);
      } else {
#pragma unroll
        for (int h = 0; h < 16; ++h) {
          const int key  = k0 + (h & 7) + g * 8 + (h >> 3) * 16;
          const int keyc = key < 0 ? 0 : (key >= Sn ? Sn - 1 : key);
          kvop[h] = kd[keyc];
        }
      }
      at[nt] = WMMA_BF16(sa, kvop, at[nt]);
    }
  }

  // ---- out = attended @ Wo + bo (attended C->A transpose via LDS) ----
#pragma unroll
  for (int nt = 0; nt < 4; ++nt)
#pragma unroll
    for (int r = 0; r < 8; ++r)
      scr[wave][r + 8 * g][nt * 16 + m] = at[nt][r];

  v16bf aa[2];
#pragma unroll
  for (int p = 0; p < 2; ++p) {
    const float* ar = &scr[wave][m][p * 32 + g * 8];
    v4f a0 = *(const v4f*)(ar),      a1 = *(const v4f*)(ar + 4);
    v4f a2 = *(const v4f*)(ar + 16), a3 = *(const v4f*)(ar + 20);
#pragma unroll
    for (int h = 0; h < 4; ++h) {
      aa[p][h] = f2bf(a0[h]); aa[p][h + 4] = f2bf(a1[h]);
      aa[p][h + 8] = f2bf(a2[h]); aa[p][h + 12] = f2bf(a3[h]);
    }
  }

  for (int nt = 0; nt < Dn / 16; ++nt) {    // 32 output column tiles
    v8f oc = {};
#pragma unroll
    for (int p = 0; p < 2; ++p) {
      const v16bf wb = ld_b_op(WoT + (size_t)(nt * 16 + m) * Dkn + p * 32 + g * 8);
      oc = WMMA_BF16(aa[p], wb, oc);
    }
    const float bv = bo[nt * 16 + m];
#pragma unroll
    for (int r = 0; r < 8; ++r)
      out[((size_t)b * Sn + q0 + r + 8 * g) * (size_t)Dn + nt * 16 + m] = oc[r] + bv;
  }
}

// ---------------------------------------------------------------------------
extern "C" void kernel_launch(void* const* d_in, const int* in_sizes, int n_in,
                              void* d_out, int out_size, void* d_ws, size_t ws_size,
                              hipStream_t stream) {
  (void)in_sizes; (void)n_in; (void)out_size; (void)ws_size;
  const float* query = (const float*)d_in[0];
  const float* value = (const float*)d_in[1];
  const float* Wi    = (const float*)d_in[2];
  const float* bi    = (const float*)d_in[3];
  const float* Wo    = (const float*)d_in[4];
  const float* bo    = (const float*)d_in[5];
  const int*   band  = (const int*)d_in[6];

  // ws layout: qB (2MB) | kvB (2MB) | kvT (2MB) | WiT (64KB) | WoT (64KB)
  char* ws = (char*)d_ws;
  const size_t qk = (size_t)Bn * Sn * Dkn;        // bf16 elements per tensor
  __bf16* qB  = (__bf16*)(ws);
  __bf16* kvB = (__bf16*)(ws + qk * 2);
  __bf16* kvT = (__bf16*)(ws + 2 * qk * 2);
  __bf16* WiT = (__bf16*)(ws + 3 * qk * 2);
  __bf16* WoT = (__bf16*)(ws + 3 * qk * 2 + (size_t)Dn * Dkn * 2);

  hipLaunchKernelGGL(sa_prep, dim3(128), dim3(256), 0, stream, Wi, Wo, WiT, WoT);
  hipLaunchKernelGGL(sa_proj, dim3(Bn * Sn / 128, 2), dim3(128), 0, stream,
                     query, value, WiT, bi, qB, kvB, kvT);
  hipLaunchKernelGGL(sa_attn, dim3(Bn * Sn / 64), dim3(128), 0, stream,
                     qB, kvB, kvT, WoT, bo, band, (float*)d_out);
}
